// SelfAttention_867583394603
// MI455X (gfx1250) — compile-verified, matrix operand order checked by reading, not run on hardware
//
#include <hip/hip_runtime.h>
#include <hip/hip_bf16.h>

typedef unsigned short u16;
typedef __attribute__((ext_vector_type(16))) __bf16 v16bf;
typedef __attribute__((ext_vector_type(8)))  __bf16 v8bf;
typedef __attribute__((ext_vector_type(8)))  float  v8f;
typedef __attribute__((ext_vector_type(4)))  int    v4i;

#define BM 128
#define BN 128
#define BKK 64
#define NTHREADS 256
#define SEQ 2048

// ---------------- CDNA5 async global->LDS path (guarded) ----------------
#if defined(__gfx1250__) && __has_builtin(__builtin_amdgcn_global_load_async_to_lds_b128)
#define HAVE_ASYNC 1
#else
#define HAVE_ASYNC 0
#endif

__device__ inline void wait_async_all() {
#if HAVE_ASYNC
#if __has_builtin(__builtin_amdgcn_s_wait_asynccnt)
    __builtin_amdgcn_s_wait_asynccnt(0);
#else
    asm volatile("s_wait_asynccnt 0" ::: "memory");
#endif
#endif
}

#if HAVE_ASYNC
__device__ inline void async_cp16B(u16* lds_dst, const u16* gsrc) {
    // global_load_async_to_lds_b128 : 16 bytes per lane, tracked by ASYNCcnt
    __builtin_amdgcn_global_load_async_to_lds_b128(
        (v4i*)gsrc, (v4i*)lds_dst, 0, 0);
}
#endif

// ---------- scalar conversions (round-to-nearest-even f32 -> bf16) ----------
__device__ inline u16 f2bf(float f) {
    union { float f; unsigned u; } x; x.f = f;
    unsigned u = x.u;
    unsigned r = u + 0x7FFFu + ((u >> 16) & 1u);
    return (u16)(r >> 16);
}
__device__ inline float bf2f(u16 b) {
    union { unsigned u; float f; } x; x.u = ((unsigned)b) << 16;
    return x.f;
}

// ---------- 16-element row loaders into LDS (bf16 raw bits) ----------
template <typename T> __device__ inline void load16(u16* dst, const T* src);

template <> __device__ inline void load16<float>(u16* dst, const float* src) {
#pragma unroll
    for (int q = 0; q < 4; ++q) {
        float4 f = *(const float4*)(src + q * 4);
        dst[q * 4 + 0] = f2bf(f.x);
        dst[q * 4 + 1] = f2bf(f.y);
        dst[q * 4 + 2] = f2bf(f.z);
        dst[q * 4 + 3] = f2bf(f.w);
    }
}
template <> __device__ inline void load16<u16>(u16* dst, const u16* src) {
    *(uint4*)(dst)     = *(const uint4*)(src);
    *(uint4*)(dst + 8) = *(const uint4*)(src + 8);
}

template <typename T> __device__ inline void store_elem(T* p, float v);
template <> __device__ inline void store_elem<float>(float* p, float v) { *p = v; }
template <> __device__ inline void store_elem<u16>(u16* p, float v) { *p = f2bf(v); }

// ---- stage a [BM][BKK] tile whose source rows are contiguous in K ----
template <typename T>
__device__ inline void stage_rows(u16* dst, const T* src, int ld, int k0, int tid) {
#if HAVE_ASYNC
    if constexpr (sizeof(T) == 2) {
        for (int i = tid * 8; i < BM * BKK; i += NTHREADS * 8) {
            int r = i / BKK, c = i % BKK;
            async_cp16B(dst + r * BKK + c,
                        (const u16*)src + (long long)r * ld + k0 + c);
        }
        return;
    }
#endif
    for (int i = tid * 16; i < BM * BKK; i += NTHREADS * 16) {
        int r = i / BKK, c = i % BKK;
        load16<T>(dst + r * BKK + c, src + (long long)r * ld + k0 + c);
    }
}

// ---- stage B [K][N] row-major -> LDS [n][k] (transpose), vectorized ----
template <typename T>
__device__ inline void stage_transpose(u16* dst, const T* src, int N, int k0, int tid) {
    for (int j = tid * 4; j < BKK * BN; j += NTHREADS * 4) {
        int kk = j / BN, n = j - kk * BN;          // n is a multiple of 4
        const T* s = src + (long long)(k0 + kk) * N + n;
        if constexpr (sizeof(T) == 4) {
            float4 f = *(const float4*)s;
            dst[(n + 0) * BKK + kk] = f2bf(f.x);
            dst[(n + 1) * BKK + kk] = f2bf(f.y);
            dst[(n + 2) * BKK + kk] = f2bf(f.z);
            dst[(n + 3) * BKK + kk] = f2bf(f.w);
        } else {
            struct U4 { u16 a, b, c, d; };
            U4 t = *(const U4*)s;
            dst[(n + 0) * BKK + kk] = t.a;
            dst[(n + 1) * BKK + kk] = t.b;
            dst[(n + 2) * BKK + kk] = t.c;
            dst[(n + 3) * BKK + kk] = t.d;
        }
    }
}

// ---------- WMMA fragment load (CDNA5 ISA 7.12.2 16-bit A/B layout) ----------
__device__ inline v16bf load_frag(const u16* p) {
    union { v16bf v; v8bf h[2]; } f;
    f.h[0] = *(const v8bf*)(p);
    f.h[1] = *(const v8bf*)(p + 16);
    return f.v;
}

// ---------- tiled, double-buffered WMMA GEMM ----------
// C[M,N] = (A[M,K] x B) * scale (+bias)
// TRANSB==true : B is [N,K] row-major   (Q@K^T)
// TRANSB==false: B is [K,N] row-major   (weights, V)
template <typename AT, typename BT, bool TRANSB, typename OT, bool BIAS>
__global__ void __launch_bounds__(NTHREADS)
gemm_wmma(const AT* __restrict__ A, const BT* __restrict__ Bm,
          const float* __restrict__ bias, OT* __restrict__ C,
          int M, int N, int K,
          long long sA, long long sB, long long sC, float scale)
{
    __shared__ u16 As[2][BM * BKK];
    __shared__ u16 Bs[2][BN * BKK];

    const int tid  = threadIdx.x;
    const int lane = tid & 31;
    const int wave = tid >> 5;              // 8 wave32s, each owns 16 rows

    const int bz = blockIdx.z;
    A  += bz * sA;
    Bm += bz * sB;
    C  += bz * sC;

    const int rowBase = blockIdx.y * BM;
    const int colBase = blockIdx.x * BN;
    const AT* Arow = A + (long long)rowBase * K;

    v8f acc[8] = {};

    // prologue: stage first tiles
    stage_rows<AT>(As[0], Arow, K, 0, tid);
    if constexpr (TRANSB)
        stage_rows<BT>(Bs[0], Bm + (long long)colBase * K, K, 0, tid);
    else
        stage_transpose<BT>(Bs[0], Bm + colBase, N, 0, tid);
    wait_async_all();
    __syncthreads();

    const int hh = lane >> 4;
    const int ml = lane & 15;
    int cur = 0;

    for (int k0 = 0; k0 < K; k0 += BKK) {
        const int nxt = cur ^ 1;
        // stage next tile while computing on current
        if (k0 + BKK < K) {
            stage_rows<AT>(As[nxt], Arow, K, k0 + BKK, tid);
            if constexpr (TRANSB)
                stage_rows<BT>(Bs[nxt], Bm + (long long)colBase * K, K, k0 + BKK, tid);
            else
                stage_transpose<BT>(Bs[nxt], Bm + colBase, N, k0 + BKK, tid);
        }

        const u16* At = As[cur];
        const u16* Bt = Bs[cur];
#pragma unroll
        for (int ks = 0; ks < BKK / 32; ++ks) {
            v16bf a = load_frag(&At[(wave * 16 + ml) * BKK + ks * 32 + hh * 8]);
            v16bf b[8];
#pragma unroll
            for (int nt = 0; nt < 8; ++nt)
                b[nt] = load_frag(&Bt[(nt * 16 + ml) * BKK + ks * 32 + hh * 8]);
#pragma unroll
            for (int nt = 0; nt < 8; ++nt)
                acc[nt] = __builtin_amdgcn_wmma_f32_16x16x32_bf16(
                    false, a, false, b[nt], (short)0, acc[nt], false, false);
        }

        wait_async_all();
        __syncthreads();
        cur = nxt;
    }

    // ---- epilogue: C/D layout — VGPR r: M = r + (lane>>4)*8, N = lane&15 ----
#pragma unroll
    for (int nt = 0; nt < 8; ++nt) {
        int n = colBase + nt * 16 + ml;
        float bb = BIAS ? bias[n] : 0.0f;
#pragma unroll
        for (int r = 0; r < 8; ++r) {
            int m = rowBase + wave * 16 + r + hh * 8;
            float val = acc[nt][r] * scale + bb;
            store_elem<OT>(&C[(long long)m * N + n], val);
        }
    }
}

// ---------- row softmax over bf16 scores, in place; row length SEQ ----------
__global__ void __launch_bounds__(NTHREADS)
softmax_rows(u16* __restrict__ P)
{
    __shared__ float red[8];
    const long long row = blockIdx.x;
    u16* p = P + row * (long long)SEQ;
    const int tid  = threadIdx.x;
    const int lane = tid & 31;
    const int wave = tid >> 5;

    float v[SEQ / NTHREADS];
    float mx = -3.4e38f;
#pragma unroll
    for (int i = 0; i < SEQ / NTHREADS; ++i) {
        v[i] = bf2f(p[tid + i * NTHREADS]);
        mx = fmaxf(mx, v[i]);
    }
#pragma unroll
    for (int off = 16; off > 0; off >>= 1)
        mx = fmaxf(mx, __shfl_xor(mx, off));
    if (lane == 0) red[wave] = mx;
    __syncthreads();
    float rowmax = red[0];
#pragma unroll
    for (int w = 1; w < 8; ++w) rowmax = fmaxf(rowmax, red[w]);
    __syncthreads();

    float sum = 0.0f;
#pragma unroll
    for (int i = 0; i < SEQ / NTHREADS; ++i) {
        v[i] = __expf(v[i] - rowmax);
        sum += v[i];
    }
#pragma unroll
    for (int off = 16; off > 0; off >>= 1)
        sum += __shfl_xor(sum, off);
    if (lane == 0) red[wave] = sum;
    __syncthreads();
    float rowsum = 0.0f;
#pragma unroll
    for (int w = 0; w < 8; ++w) rowsum += red[w];
    float inv = 1.0f / rowsum;

#pragma unroll
    for (int i = 0; i < SEQ / NTHREADS; ++i)
        p[tid + i * NTHREADS] = f2bf(v[i] * inv);
}

// ---------- launch ----------
extern "C" void kernel_launch(void* const* d_in, const int* in_sizes, int n_in,
                              void* d_out, int out_size, void* d_ws, size_t ws_size,
                              hipStream_t stream)
{
    (void)in_sizes; (void)n_in; (void)out_size; (void)ws_size;
    constexpr int D = 1024, S = SEQ, B = 4;
    constexpr float SCALE = 0.03125f;  // 1/sqrt(1024)

    const float* x  = (const float*)d_in[0];
    const float* Wq = (const float*)d_in[1];
    const float* bq = (const float*)d_in[2];
    const float* Wk = (const float*)d_in[3];
    const float* bk = (const float*)d_in[4];
    const float* Wv = (const float*)d_in[5];
    const float* bv = (const float*)d_in[6];
    const float* Wo = (const float*)d_in[7];
    const float* bo = (const float*)d_in[8];
    float* out = (float*)d_out;

    // workspace layout (bf16 buffers)
    u16* qb = (u16*)d_ws;                      // [B*S, D]
    u16* kb = qb + (size_t)B * S * D;          // [B*S, D]
    u16* vb = kb + (size_t)B * S * D;          // [B*S, D]
    u16* pb = vb + (size_t)B * S * D;          // [B, S, S] scores -> probs
    u16* ab = pb + (size_t)B * S * S;          // [B*S, D] attention output

    dim3 blk(NTHREADS);

    // 1) QKV projections: [8192,1024] = x @ W + b  (bf16 out)
    dim3 g1(D / BN, (B * S) / BM);
    gemm_wmma<float, float, false, u16, true><<<g1, blk, 0, stream>>>(
        x, Wq, bq, qb, B * S, D, D, 0, 0, 0, 1.0f);
    gemm_wmma<float, float, false, u16, true><<<g1, blk, 0, stream>>>(
        x, Wk, bk, kb, B * S, D, D, 0, 0, 0, 1.0f);
    gemm_wmma<float, float, false, u16, true><<<g1, blk, 0, stream>>>(
        x, Wv, bv, vb, B * S, D, D, 0, 0, 0, 1.0f);

    // 2) scores = Q @ K^T * scale   (batched over B, bf16 out, L2-resident)
    dim3 g2(S / BN, S / BM, B);
    gemm_wmma<u16, u16, true, u16, false><<<g2, blk, 0, stream>>>(
        qb, kb, nullptr, pb, S, S, D,
        (long long)S * D, (long long)S * D, (long long)S * S, SCALE);

    // 3) softmax over each score row (in place)
    softmax_rows<<<B * S, NTHREADS, 0, stream>>>(pb);

    // 4) att = P @ V   (batched)
    dim3 g3(D / BN, S / BM, B);
    gemm_wmma<u16, u16, false, u16, false><<<g3, blk, 0, stream>>>(
        pb, vb, nullptr, ab, S, D, S,
        (long long)S * S, (long long)S * D, (long long)S * D, 1.0f);

    // 5) out = att @ Wo + bo   (fp32 out)
    dim3 g4(D / BN, (B * S) / BM);
    gemm_wmma<u16, float, false, float, true><<<g4, blk, 0, stream>>>(
        ab, Wo, bo, out, B * S, D, D, 0, 0, 0, 1.0f);
}
